// Attention_48790828483196
// MI455X (gfx1250) — compile-verified, matrix operand order checked by reading, not run on hardware
//
#include <hip/hip_runtime.h>

// ---------------------------------------------------------------------------
// Causal attention, B=4, S=2048, d_model=d_k=d_v=1024, fp32 in/out.
// bf16 WMMA pipeline with TDM (tensor_load_to_lds) double-buffered LDS staging.
//   1) convert x, Wq, Wk, Wv  f32 -> bf16
//   2) Q = x Wq^T, K = x Wk^T, Vt = (x Wv^T)^T      (128x128 block GEMM)
//   3) scores = (Q K^T)/32, causal (skip blocks above diagonal)
//   4) row-wise masked softmax -> P (bf16, zeros above diagonal)
//   5) out = P * Vt^T, K-loop truncated at the diagonal
// Block GEMM: 512 threads = 16 waves (8 along M x 2 along N), wave tile 16x64
// (4 accumulators = 32 VGPRs) to keep accumulators resident (no spills).
// ---------------------------------------------------------------------------

typedef __attribute__((ext_vector_type(16))) __bf16        v16bf;
typedef __attribute__((ext_vector_type(8)))  __bf16        v8bf;
typedef __attribute__((ext_vector_type(8)))  float         v8f;
typedef __attribute__((ext_vector_type(4)))  unsigned int  u32x4;
typedef __attribute__((ext_vector_type(8)))  int           i32x8;
typedef __attribute__((ext_vector_type(4)))  int           i32x4;

#define BATCH    4
#define SEQ      2048
#define DMODEL   1024
#define SCALE_QK 0.03125f    // 1/sqrt(1024)

// TDM pads 4 DWORDs (16B) after every 64B line -> 80B LDS pitch (40 bf16).
// Keeps ds_load_b128 16B-aligned; 16 consecutive rows hit distinct banks.
#define LDS_PITCH 40

// ---- LDS byte offset of a __shared__ object (flat addr low 32 bits) -----
__device__ __forceinline__ unsigned lds_off_of(const void* p) {
  return (unsigned)(unsigned long long)(size_t)p;
}

// ---- TDM descriptor group 1 (invariant per GEMM): data_size=2B, pad 16B
// per 64B line, tile 32 x 128, line stride = ld_elems ----------------------
__device__ __forceinline__ i32x8 tdm_group1(int ld_elems) {
  const unsigned dim0 = (unsigned)ld_elems;     // line length (elements)
  const unsigned dim1 = 1u << 20;               // plenty of rows (in-bounds)
  i32x8 g1;
  g1[0] = (int)((1u << 16)        // data_size = 1 -> 2 bytes
              | (1u << 20)        // pad_enable
              | (3u << 22)        // pad_interval = 3 -> every 16 DWORDs (64B)
              | (3u << 25));      // pad_amount  = 3 -> 4 DWORDs (16B)
  g1[1] = (int)(dim0 << 16);                    // tensor_dim0[15:0]
  g1[2] = (int)((dim0 >> 16) | (dim1 << 16));   // dim0 hi | dim1 lo
  g1[3] = (int)((dim1 >> 16) | (32u << 16));    // dim1 hi | tile_dim0=32
  g1[4] = 128;                                  // tile_dim1=128, tile_dim2=0
  g1[5] = ld_elems;                             // tensor_dim0_stride lo
  g1[6] = 0;
  g1[7] = 0;
  return g1;
}

// ---- TDM issue: group 0 carries the per-chunk addresses ------------------
__device__ __forceinline__ void tdm_issue(i32x8 g1, const __bf16* gptr,
                                          unsigned lds_off) {
  unsigned long long ga = (unsigned long long)(size_t)gptr;
  u32x4 g0;
  g0[0] = 1u;                                   // count=1 (valid), user mode
  g0[1] = lds_off;                              // lds_addr
  g0[2] = (unsigned)ga;                         // global_addr[31:0]
  g0[3] = (unsigned)(ga >> 32) | (2u << 30);    // global_addr[56:32] | type=2
  i32x4 z4 = {0, 0, 0, 0};                      // groups 2/3 unused (2-D tile)
  i32x8 z8 = {0, 0, 0, 0, 0, 0, 0, 0};
  __builtin_amdgcn_tensor_load_to_lds(g0, g1, z4, z4, z8, 0);
}

// ---- fragment gather from an LDS tile row -------------------------------
// 16-bit A/B fragment layout (ISA 7.12.2): lane&15 = row (A:M / B:N),
// lane>>4 selects K half {0..7,16..23} vs {8..15,24..31}. Two ds_load_b128.
__device__ __forceinline__ v16bf load_frag(const __bf16* p, int hl) {
  v8bf lo = *reinterpret_cast<const v8bf*>(p + hl * 8);
  v8bf hi = *reinterpret_cast<const v8bf*>(p + 16 + hl * 8);
  return __builtin_shufflevector(lo, hi, 0, 1, 2, 3, 4, 5, 6, 7,
                                         8, 9, 10, 11, 12, 13, 14, 15);
}

__device__ __forceinline__ v8f wmma_bf16(v16bf a, v16bf b, v8f c) {
  return __builtin_amdgcn_wmma_f32_16x16x32_bf16(false, a, false, b,
                                                 (short)0, c, false, false);
}

// ---- block GEMM: C[128x128] = A[128xK] * B[128xK]^T ---------------------
// 512 threads = 16 waves (8 M x 2 N); wave tile 16x64 = 4 WMMA tiles.
// TDM double-buffers 128x32 K-chunks of A and B in LDS.
__device__ __forceinline__ void block_gemm_128x128(
    const __bf16* __restrict__ A, int lda,
    const __bf16* __restrict__ B, int ldb,
    int K, v8f c[4])
{
  __shared__ __bf16 sA[2][128 * LDS_PITCH];
  __shared__ __bf16 sB[2][128 * LDS_PITCH];

  const int tid  = threadIdx.x;
  const int wv   = tid >> 5;
  const int lane = tid & 31;
  const int hl   = lane >> 4;
  const int r    = lane & 15;
  const int wm   = (wv >> 1) * 16;    // wave rows: 0..112 step 16
  const int wn   = (wv & 1) * 64;     // wave cols: 0 / 64
  const int nK   = K >> 5;

  const unsigned offA0 = lds_off_of(&sA[0][0]);
  const unsigned offA1 = lds_off_of(&sA[1][0]);
  const unsigned offB0 = lds_off_of(&sB[0][0]);
  const unsigned offB1 = lds_off_of(&sB[1][0]);

  const i32x8 g1a = tdm_group1(lda);
  const i32x8 g1b = tdm_group1(ldb);

  if (wv == 0) {                      // prologue: stage chunk 0 (wave 0 only)
    tdm_issue(g1a, A, offA0);
    tdm_issue(g1b, B, offB0);
  }

  for (int kc = 0; kc < nK; ++kc) {
    const int cur = kc & 1;
    if (wv == 0) {
      if (kc + 1 < nK) {              // stage next chunk into other buffer
        const int k = (kc + 1) << 5;
        tdm_issue(g1a, A + k, cur ? offA0 : offA1);
        tdm_issue(g1b, B + k, cur ? offB0 : offB1);
        __builtin_amdgcn_s_wait_tensorcnt(2);   // oldest 2 (chunk kc) done
      } else {
        __builtin_amdgcn_s_wait_tensorcnt(0);
      }
    }
    __syncthreads();                  // chunk kc visible to all waves

    const __bf16* ap = &sA[cur][(wm + r) * LDS_PITCH];
    const __bf16* bp = &sB[cur][(wn + r) * LDS_PITCH];
    v16bf fa = load_frag(ap, hl);
#pragma unroll
    for (int t = 0; t < 4; ++t) {
      v16bf fb = load_frag(bp + t * 16 * LDS_PITCH, hl);
      c[t] = wmma_bf16(fa, fb, c[t]);
    }
    __syncthreads();                  // done reading buf[cur]; safe to overwrite
  }
}

// ---- 1) f32 -> bf16 conversion ------------------------------------------
__global__ void cvt_f32_to_bf16(const float* __restrict__ src,
                                __bf16* __restrict__ dst, int n) {
  for (int i = blockIdx.x * blockDim.x + threadIdx.x; i < n;
       i += gridDim.x * blockDim.x)
    dst[i] = (__bf16)src[i];
}

// ---- 2) projection: Out[m,n] = sum_k X[m,k] W[n,k] ----------------------
// grid(8, 64): M = 8192, N = 1024. transposed=1 stores Vt[b, n, s].
__global__ void __launch_bounds__(512)
proj_gemm(const __bf16* __restrict__ X,
          const __bf16* __restrict__ W,
          __bf16* __restrict__ Out, int transposed) {
  const int m0 = blockIdx.y * 128, n0 = blockIdx.x * 128;
  const int wv = threadIdx.x >> 5, lane = threadIdx.x & 31;
  const int hl = lane >> 4, r = lane & 15;
  const int wm = (wv >> 1) * 16, wn = (wv & 1) * 64;

  v8f c[4] = {};
  block_gemm_128x128(X + (size_t)m0 * DMODEL, DMODEL,
                     W + (size_t)n0 * DMODEL, DMODEL, DMODEL, c);

#pragma unroll
  for (int t = 0; t < 4; ++t)
#pragma unroll
    for (int e = 0; e < 8; ++e) {
      const int row = m0 + wm + hl * 8 + e;     // token (b*SEQ+s)
      const int col = n0 + wn + t * 16 + r;     // feature
      const __bf16 v = (__bf16)c[t][e];
      if (!transposed) {
        Out[(size_t)row * DMODEL + col] = v;
      } else {
        const int b = row >> 11, s = row & (SEQ - 1);
        Out[((size_t)b * DMODEL + col) * SEQ + s] = v;
      }
    }
}

// ---- 3) causal scores: Sc[b,i,j] = scale * <Q_i, K_j> -------------------
// grid(16, 16, BATCH). Blocks fully above the diagonal exit early (uniform).
__global__ void __launch_bounds__(512)
scores_gemm(const __bf16* __restrict__ Q,
            const __bf16* __restrict__ Kmat,
            float* __restrict__ Sc) {
  if (blockIdx.x > blockIdx.y) return;          // whole 128x128 block masked
  const int b = blockIdx.z;
  const int m0 = blockIdx.y * 128, n0 = blockIdx.x * 128;
  const int wv = threadIdx.x >> 5, lane = threadIdx.x & 31;
  const int hl = lane >> 4, r = lane & 15;
  const int wm = (wv >> 1) * 16, wn = (wv & 1) * 64;

  v8f c[4] = {};
  block_gemm_128x128(Q    + ((size_t)b * SEQ + m0) * DMODEL, DMODEL,
                     Kmat + ((size_t)b * SEQ + n0) * DMODEL, DMODEL, DMODEL, c);

#pragma unroll
  for (int t = 0; t < 4; ++t)
#pragma unroll
    for (int e = 0; e < 8; ++e) {
      const int row = m0 + wm + hl * 8 + e;     // query i
      const int col = n0 + wn + t * 16 + r;     // key j
      const float v = (col <= row) ? c[t][e] * SCALE_QK : -__builtin_inff();
      Sc[((size_t)b * SEQ + row) * SEQ + col] = v;
    }
}

// ---- 4) masked row softmax -> bf16 P (zeros above diagonal) -------------
__global__ void softmax_rows(const float* __restrict__ Sc,
                             __bf16* __restrict__ P) {
  const int rowid = blockIdx.x;                 // b*SEQ + i
  const int i = rowid & (SEQ - 1);
  const int L = i + 1;
  const float* srow = Sc + (size_t)rowid * SEQ;
  __bf16* prow      = P  + (size_t)rowid * SEQ;
  __shared__ float red[256];
  const int tid = threadIdx.x;

  float m = -__builtin_inff();
  for (int j = tid; j < L; j += 256) m = fmaxf(m, srow[j]);
  red[tid] = m; __syncthreads();
  for (int s = 128; s > 0; s >>= 1) {
    if (tid < s) red[tid] = fmaxf(red[tid], red[tid + s]);
    __syncthreads();
  }
  m = red[0]; __syncthreads();

  float sum = 0.f;
  for (int j = tid; j < L; j += 256) sum += __expf(srow[j] - m);
  red[tid] = sum; __syncthreads();
  for (int s = 128; s > 0; s >>= 1) {
    if (tid < s) red[tid] += red[tid + s];
    __syncthreads();
  }
  const float inv = 1.0f / red[0];

  for (int j = tid; j < SEQ; j += 256)
    prow[j] = (j < L) ? (__bf16)(__expf(srow[j] - m) * inv) : (__bf16)0.0f;
}

// ---- 5) out[b,i,v] = sum_j P[b,i,j] Vt[b,v,j] ---------------------------
// grid(8, 16, BATCH). P rows have zeros past the diagonal -> K = m0+128.
__global__ void __launch_bounds__(512)
pv_gemm(const __bf16* __restrict__ P,
        const __bf16* __restrict__ Vt,
        float* __restrict__ Out) {
  const int b = blockIdx.z;
  const int m0 = blockIdx.y * 128, n0 = blockIdx.x * 128;
  const int wv = threadIdx.x >> 5, lane = threadIdx.x & 31;
  const int hl = lane >> 4, r = lane & 15;
  const int wm = (wv >> 1) * 16, wn = (wv & 1) * 64;
  const int Keff = m0 + 128;                    // multiple of 32

  v8f c[4] = {};
  block_gemm_128x128(P  + ((size_t)b * SEQ + m0) * SEQ, SEQ,
                     Vt + ((size_t)b * DMODEL + n0) * SEQ, SEQ, Keff, c);

#pragma unroll
  for (int t = 0; t < 4; ++t)
#pragma unroll
    for (int e = 0; e < 8; ++e) {
      const int row = m0 + wm + hl * 8 + e;
      const int col = n0 + wn + t * 16 + r;
      Out[((size_t)b * SEQ + row) * DMODEL + col] = c[t][e];
    }
}

// ---------------------------------------------------------------------------
extern "C" void kernel_launch(void* const* d_in, const int* in_sizes, int n_in,
                              void* d_out, int out_size, void* d_ws, size_t ws_size,
                              hipStream_t stream) {
  const float* x  = (const float*)d_in[0];
  // d_in[1] = mask (causal, implicit) -- unused
  const float* Wq = (const float*)d_in[2];
  const float* Wk = (const float*)d_in[3];
  const float* Wv = (const float*)d_in[4];
  float* out = (float*)d_out;

  const size_t nX = (size_t)BATCH * SEQ * DMODEL;   // 8388608
  const size_t nW = (size_t)DMODEL * DMODEL;        // 1048576
  const size_t nS = (size_t)BATCH * SEQ * SEQ;      // 16777216

  char* ws = (char*)d_ws;
  size_t off = 0;
  __bf16* xb  = (__bf16*)(ws + off); off += nX * 2;     // 16 MB
  __bf16* wqb = (__bf16*)(ws + off); off += nW * 2;     //  2 MB
  __bf16* wkb = (__bf16*)(ws + off); off += nW * 2;     //  2 MB
  __bf16* wvb = (__bf16*)(ws + off); off += nW * 2;     //  2 MB
  __bf16* Qb  = (__bf16*)(ws + off); off += nX * 2;     // 16 MB
  __bf16* Kb  = (__bf16*)(ws + off); off += nX * 2;     // 16 MB
  __bf16* Vtb = (__bf16*)(ws + off); off += nX * 2;     // 16 MB
  float*  Sc  = (float*) (ws + off); off += nS * 4;     // 64 MB
  __bf16* Pb  = (__bf16*)(ws + off); off += nS * 2;     // 32 MB

  // 1) f32 -> bf16
  cvt_f32_to_bf16<<<4096, 256, 0, stream>>>(x,  xb,  (int)nX);
  cvt_f32_to_bf16<<<1024, 256, 0, stream>>>(Wq, wqb, (int)nW);
  cvt_f32_to_bf16<<<1024, 256, 0, stream>>>(Wk, wkb, (int)nW);
  cvt_f32_to_bf16<<<1024, 256, 0, stream>>>(Wv, wvb, (int)nW);

  // 2) projections; V stored transposed per batch
  dim3 pg(DMODEL / 128, (BATCH * SEQ) / 128);
  proj_gemm<<<pg, 512, 0, stream>>>(xb, wqb, Qb,  0);
  proj_gemm<<<pg, 512, 0, stream>>>(xb, wkb, Kb,  0);
  proj_gemm<<<pg, 512, 0, stream>>>(xb, wvb, Vtb, 1);

  // 3) causal scores
  scores_gemm<<<dim3(SEQ / 128, SEQ / 128, BATCH), 512, 0, stream>>>(Qb, Kb, Sc);

  // 4) softmax rows
  softmax_rows<<<BATCH * SEQ, 256, 0, stream>>>(Sc, Pb);

  // 5) P @ V
  pv_gemm<<<dim3(DMODEL / 128, SEQ / 128, BATCH), 512, 0, stream>>>(Pb, Vtb, out);
}
